// WMSA_16518444220926
// MI455X (gfx1250) — matrix operand order repaired and checked by reading, not compiled, not run on hardware
//
#include <hip/hip_runtime.h>

// ---------------------------------------------------------------------------
// Fused shifted-window MSA for MI455X (gfx1250), bf16 WMMA path.
// ---------------------------------------------------------------------------

typedef __attribute__((ext_vector_type(16))) __bf16 v16bf;
typedef __attribute__((ext_vector_type(8)))  __bf16 v8bf;
typedef __attribute__((ext_vector_type(8)))  float  v8f;

__device__ __forceinline__ v8f wmma_bf16(v16bf a, v16bf b, v8f c) {
    // D = A(16x32) * B(32x16) + C, f32 accumulate
    return __builtin_amdgcn_wmma_f32_16x16x32_bf16(
        /*neg_a=*/false, a, /*neg_b=*/false, b,
        /*c_mod=*/(short)0, c, /*reuse_a=*/false, /*reuse_b=*/false);
}

// A-fragment: row-major matrix M[row][col], element (m,k) in contiguous k order:
// lane m = lane&15, half = lane>>4; a[i] covers k = half*16 + i.
// Two contiguous 16B chunks per lane.
__device__ __forceinline__ v16bf load_a(const __bf16* base, int stride,
                                        int row0, int col0, int lane) {
    int m = lane & 15, half = lane >> 4;
    const __bf16* p = base + (row0 + m) * stride + col0 + half * 16;
    v8bf lo = *(const v8bf*)(p);        // k = half*16 + [0..7]
    v8bf hi = *(const v8bf*)(p + 8);    // k = half*16 + 8 + [0..7]
    return __builtin_shufflevector(lo, hi, 0, 1, 2, 3, 4, 5, 6, 7,
                                           8, 9, 10, 11, 12, 13, 14, 15);
}

// B-fragment: element (k,n) stored at base[(n0+n)*stride + k0 + k]
// lane n = lane&15, half = lane>>4; b[i] covers k = half*16 + i.
// One contiguous 32B chunk per lane.
__device__ __forceinline__ v16bf load_b(const __bf16* base, int stride,
                                        int n0, int k0, int lane) {
    int n = lane & 15, half = lane >> 4;
    return *(const v16bf*)(base + (n0 + n) * stride + k0 + half * 16);
}

// ---------------------------------------------------------------------------
// Prep: convert W_qkv (192x576) and W_out (192x192) fp32 -> bf16, pre-swizzled
// into per-(ntile,kblock) lane-major B-fragment layout:
//   sw[((tn*6+kb)*32 + lane)*16 + i] = W[(kb*32 + (lane>>4)*16 + i)][tn*16 + (lane&15)]
// ---------------------------------------------------------------------------
__global__ void prep_weights(const float* __restrict__ Wqkv,
                             const float* __restrict__ Wout,
                             __bf16* __restrict__ wq_sw,
                             __bf16* __restrict__ wo_sw) {
    const int N1 = 36 * 6 * 32 * 16;   // 110592  (W_qkv: 36 n-tiles)
    const int N2 = 12 * 6 * 32 * 16;   // 36864   (W_out: 12 n-tiles)
    int idx = blockIdx.x * blockDim.x + threadIdx.x;
    if (idx < N1) {
        int i    = idx & 15;
        int lane = (idx >> 4) & 31;
        int kb   = (idx >> 9) % 6;
        int tn   = idx / (6 * 512);
        int k = kb * 32 + (lane >> 4) * 16 + i;
        int n = tn * 16 + (lane & 15);
        wq_sw[idx] = (__bf16)Wqkv[k * 576 + n];
    } else if (idx < N1 + N2) {
        int j    = idx - N1;
        int i    = j & 15;
        int lane = (j >> 4) & 31;
        int kb   = (j >> 9) % 6;
        int tn   = j / (6 * 512);
        int k = kb * 32 + (lane >> 4) * 16 + i;
        int n = tn * 16 + (lane & 15);
        wo_sw[j] = (__bf16)Wout[k * 192 + n];
    }
}

// ---------------------------------------------------------------------------
// Fused per-window kernel. 1 block = one (batch, window). 256 threads = 8 waves.
// Dynamic LDS: 147456 bytes.
// ---------------------------------------------------------------------------
__global__ __launch_bounds__(256) void wmsa_fused(
    const float*  __restrict__ x,        // [4][256][256][192]
    const float*  __restrict__ b_qkv,    // [576]
    const float*  __restrict__ rel_pos,  // [225][6]
    const float*  __restrict__ b_out,    // [192]
    const __bf16* __restrict__ wq_sw,    // swizzled bf16 W_qkv
    const __bf16* __restrict__ wo_sw,    // swizzled bf16 W_out
    float*        __restrict__ out) {    // [4][256][256][192]
    extern __shared__ char smem[];
    __bf16* Xw  = (__bf16*)(smem);            // [64][192] window input
    __bf16* Qs  = (__bf16*)(smem + 24576);    // [64][192] Q (pre-scaled)
    __bf16* Ks  = (__bf16*)(smem + 49152);    // [64][192] K
    __bf16* Vt  = (__bf16*)(smem + 73728);    // [192][64] V transposed
    float*  Sim = (float*) (smem + 98304);    // [64][64]  scores / exp
    __bf16* Pb  = (__bf16*)(smem + 114688);   // [64][64]  probs bf16
    __bf16* Ob  = (__bf16*)(smem + 122880);   // [64][192] attention output

    const int tid  = threadIdx.x;
    const int lane = tid & 31;
    const int wv   = tid >> 5;
    const int half = lane >> 4;
    const int nlo  = lane & 15;

    const int blk  = blockIdx.x;
    const int b    = blk >> 10;        // batch
    const int widx = blk & 1023;       // window id
    const int wi = widx >> 5, wj = widx & 31;
    const bool lastR = (wi == 31), lastC = (wj == 31);

    // ---- Phase 1: gather rolled window, fp32 -> bf16 into LDS --------------
    for (int it = tid; it < 64 * 48; it += 256) {
        int tok = it / 48, g = it % 48;
        int pi = tok >> 3, pj = tok & 7;
        int gi = (wi * 8 + pi + 4) & 255;   // roll(-4) read offset
        int gj = (wj * 8 + pj + 4) & 255;
        const float4 v =
            *(const float4*)(x + (((size_t)b * 256 + gi) * 256 + gj) * 192 + g * 4);
        __bf16* d = Xw + tok * 192 + g * 4;
        d[0] = (__bf16)v.x; d[1] = (__bf16)v.y;
        d[2] = (__bf16)v.z; d[3] = (__bf16)v.w;
    }
    __syncthreads();

    // ---- Phase 2: QKV = Xw @ W_qkv + b  (144 16x16 tiles, K=192) -----------
    const float qscale = 0.17677669529663687f;  // 1/sqrt(32), folded into Q
    for (int jt = wv; jt < 144; jt += 8) {
        int tm = jt & 3, tn = jt >> 2;   // tm is constant per wave -> A frags hoisted
        v8f c = {};
        #pragma unroll
        for (int kb = 0; kb < 6; ++kb) {
            v16bf a  = load_a(Xw, 192, tm * 16, kb * 32, lane);
            v16bf bb = *(const v16bf*)(wq_sw + (tn * 6 + kb) * 512 + lane * 16);
            c = wmma_bf16(a, bb, c);
        }
        float bv = b_qkv[tn * 16 + nlo];
        // section (q/k/v) is uniform per tile: force a scalar branch
        int sec = __builtin_amdgcn_readfirstlane(tn / 12);
        int ch  = (tn % 12) * 16 + nlo;   // channel within section [0,192)
        if (sec == 0) {
            #pragma unroll
            for (int r = 0; r < 8; ++r) {
                int tok = tm * 16 + half * 8 + r;
                Qs[tok * 192 + ch] = (__bf16)((c[r] + bv) * qscale);
            }
        } else if (sec == 1) {
            #pragma unroll
            for (int r = 0; r < 8; ++r) {
                int tok = tm * 16 + half * 8 + r;
                Ks[tok * 192 + ch] = (__bf16)(c[r] + bv);
            }
        } else {
            #pragma unroll
            for (int r = 0; r < 8; ++r) {
                int tok = tm * 16 + half * 8 + r;
                Vt[ch * 64 + tok] = (__bf16)(c[r] + bv);
            }
        }
    }
    __syncthreads();

    // ---- Phase 3: attention per head --------------------------------------
    for (int h = 0; h < 6; ++h) {
        // (a) scores: sim = Qs_h @ Ks_h^T  (+ rel bias, + shift mask)
        for (int t = wv; t < 16; t += 8) {
            int tp = t >> 2, tq = t & 3;
            v16bf a  = load_a(Qs, 192, tp * 16, h * 32, lane);
            v16bf bb = load_b(Ks, 192, tq * 16, h * 32, lane);
            v8f c = {};
            c = wmma_bf16(a, bb, c);
            int q = tq * 16 + nlo;
            int qi = q >> 3, qj = q & 7;
            #pragma unroll
            for (int r = 0; r < 8; ++r) {
                int p = tp * 16 + half * 8 + r;
                int pi = p >> 3, pj = p & 7;
                int ridx = (pi - qi + 7) * 15 + (pj - qj + 7);
                float v = c[r] + rel_pos[ridx * 6 + h];
                bool m = (lastR && ((pi < 4) != (qi < 4))) ||
                         (lastC && ((pj < 4) != (qj < 4)));
                Sim[p * 64 + q] = m ? -1.0e9f : v;
            }
        }
        __syncthreads();
        // (b) row softmax -> bf16 probs
        if (tid < 64) {
            float mx = -3.4e38f;
            for (int q = 0; q < 64; ++q) mx = fmaxf(mx, Sim[tid * 64 + q]);
            float s = 0.f;
            for (int q = 0; q < 64; ++q) {
                float e = __expf(Sim[tid * 64 + q] - mx);
                Sim[tid * 64 + q] = e;
                s += e;
            }
            float inv = 1.0f / s;
            for (int q = 0; q < 64; ++q)
                Pb[tid * 64 + q] = (__bf16)(Sim[tid * 64 + q] * inv);
        }
        __syncthreads();
        // (c) out_h = P @ V_h  (8 tiles, one per wave, K=64)
        {
            int tp = wv >> 1, tn = wv & 1;
            v8f c = {};
            #pragma unroll
            for (int kb = 0; kb < 2; ++kb) {
                v16bf a  = load_a(Pb, 64, tp * 16, kb * 32, lane);
                v16bf bb = load_b(Vt, 64, h * 32 + tn * 16, kb * 32, lane);
                c = wmma_bf16(a, bb, c);
            }
            int ch = h * 32 + tn * 16 + nlo;
            #pragma unroll
            for (int r = 0; r < 8; ++r) {
                int tok = tp * 16 + half * 8 + r;
                Ob[tok * 192 + ch] = (__bf16)c[r];
            }
        }
        __syncthreads();
    }

    // ---- Phase 4: Y = Ob @ W_out + b_out, scatter with inverse roll --------
    for (int jt = wv; jt < 48; jt += 8) {
        int tm = jt & 3, tn = jt >> 2;   // tm constant per wave -> A frags hoisted
        v8f c = {};
        #pragma unroll
        for (int kb = 0; kb < 6; ++kb) {
            v16bf a  = load_a(Ob, 192, tm * 16, kb * 32, lane);
            v16bf bb = *(const v16bf*)(wo_sw + (tn * 6 + kb) * 512 + lane * 16);
            c = wmma_bf16(a, bb, c);
        }
        int ch = tn * 16 + nlo;
        float bo = b_out[ch];
        #pragma unroll
        for (int r = 0; r < 8; ++r) {
            int tok = tm * 16 + half * 8 + r;
            int pi = tok >> 3, pj = tok & 7;
            int gi = (wi * 8 + pi + 4) & 255;   // roll(+4) on write == same map
            int gj = (wj * 8 + pj + 4) & 255;
            out[(((size_t)b * 256 + gi) * 256 + gj) * 192 + ch] = c[r] + bo;
        }
    }
}

// ---------------------------------------------------------------------------
extern "C" void kernel_launch(void* const* d_in, const int* in_sizes, int n_in,
                              void* d_out, int out_size, void* d_ws, size_t ws_size,
                              hipStream_t stream) {
    const float* x    = (const float*)d_in[0];
    const float* Wqkv = (const float*)d_in[1];
    const float* bqkv = (const float*)d_in[2];
    const float* relp = (const float*)d_in[3];
    const float* Wout = (const float*)d_in[4];
    const float* bout = (const float*)d_in[5];
    float* out = (float*)d_out;

    __bf16* wq_sw = (__bf16*)d_ws;            // 110592 bf16
    __bf16* wo_sw = wq_sw + 110592;           // 36864 bf16

    // Allow >64KB dynamic LDS (gfx1250 WGP has 320KB).
    static_cast<void>(hipFuncSetAttribute(
        (const void*)wmsa_fused, hipFuncAttributeMaxDynamicSharedMemorySize, 147456));

    // weight convert+swizzle: 147456 elements total
    prep_weights<<<576, 256, 0, stream>>>(Wqkv, Wout, wq_sw, wo_sw);

    // one block per (batch, window) = 4 * 1024
    wmsa_fused<<<4096, 256, 147456, stream>>>(x, bqkv, relp, bout, wq_sw, wo_sw, out);
}
